// KPConv_71451075936921
// MI455X (gfx1250) — matrix operand order repaired and checked by reading, not compiled, not run on hardware
//
#include <hip/hip_runtime.h>
#include <hip/hip_bf16.h>

typedef __attribute__((ext_vector_type(16))) _Float16 v16h;
typedef __attribute__((ext_vector_type(8)))  float    v8f;

#define QT   32          // queries per workgroup (two 16-row WMMA M tiles)
#define KC   960         // K*C contraction depth (15*64)
#define NB   30          // 960 / 32 k-blocks
#define THREADS 256

// LDS layout (dynamic):
//   [0)        48  floats : kernel points (15*3, padded)
//   [192)   30720  floats : agg[32][960]  f32
//   [123072) 61440 halves : W in WMMA-B fragment order
#define LDS_KERN_OFF 0
#define LDS_AGG_OFF  192
#define LDS_WF_OFF   123072
#define LDS_BYTES    245952

union H2U { _Float16 h[2]; unsigned u; };

__global__ __launch_bounds__(THREADS, 1)
void kpconv_fused(const float* __restrict__ qpts,      // [Q,3]
                  const float* __restrict__ feats,     // [N,64]
                  const float* __restrict__ kern,      // [15,3]
                  const float* __restrict__ W,         // [15,64,64]
                  const int*   __restrict__ nbr,       // [Q,32]
                  float* __restrict__ out)             // [Q,64]
{
    extern __shared__ char smem[];
    float* skern = reinterpret_cast<float*>(smem + LDS_KERN_OFF * 4);
    float* agg   = reinterpret_cast<float*>(smem) + LDS_AGG_OFF;
    char*  wfrag = smem + LDS_WF_OFF * 2;   // halves region (byte base = 123072*2)

    const int tid  = threadIdx.x;
    const int lane = tid & 31;
    const int warp = tid >> 5;
    const int q0   = blockIdx.x * QT;

    // ---------------- Phase 0: stage kernel pts, zero agg, swizzle W -> LDS f16 B-fragments
    if (tid < 45) skern[tid] = kern[tid];

    {   // zero agg: 32*960 f32 = 7680 float4
        float4 z = make_float4(0.f, 0.f, 0.f, 0.f);
        float4* a4 = reinterpret_cast<float4*>(agg);
        for (int i = tid; i < (QT * KC) / 4; i += THREADS) a4[i] = z;
    }

    {   // W fragment fill: 120 (b,t) pairs, 8 warps -> 15 pairs per warp
        const int kHalf = (lane < 16) ? 0 : 16;
        const int oLane = lane & 15;
        for (int i = 0; i < 15; ++i) {
            int bt = warp + 8 * i;          // 0..119
            int b  = bt >> 2;               // k-block 0..29
            int t  = bt & 3;                // o-tile  0..3
            int kc0 = b * 32 + kHalf;
            int o   = t * 16 + oLane;
            unsigned u[8];
            #pragma unroll
            for (int e = 0; e < 16; e += 2) {
                H2U p;
                p.h[0] = (_Float16)W[(size_t)(kc0 + e    ) * 64 + o];
                p.h[1] = (_Float16)W[(size_t)(kc0 + e + 1) * 64 + o];
                u[e >> 1] = p.u;
            }
            uint4* dst = reinterpret_cast<uint4*>(wfrag + ((size_t)(bt * 32 + lane) << 5));
            dst[0] = make_uint4(u[0], u[1], u[2], u[3]);
            dst[1] = make_uint4(u[4], u[5], u[6], u[7]);
        }
    }
    __syncthreads();

    // ---------------- Phase 1: geometry + influence + scatter into agg (LDS atomics)
    #pragma unroll
    for (int j = 0; j < 4; ++j) {
        int p  = tid + THREADS * j;     // 0..1023 ; lanes of a wave share q
        int ql = p >> 5;                // local query 0..31
        int h  = p & 31;
        int q  = q0 + ql;

        int nidx = nbr[(size_t)q * 32 + h];
        float qx = qpts[(size_t)q * 3 + 0];
        float qy = qpts[(size_t)q * 3 + 1];
        float qz = qpts[(size_t)q * 3 + 2];
        float dx = qpts[(size_t)nidx * 3 + 0] - qx;
        float dy = qpts[(size_t)nidx * 3 + 1] - qy;
        float dz = qpts[(size_t)nidx * 3 + 2] - qz;

        float best = 3.4e38f; int kb = 0;
        #pragma unroll
        for (int k = 0; k < 15; ++k) {
            float ex = dx - skern[k * 3 + 0];
            float ey = dy - skern[k * 3 + 1];
            float ez = dz - skern[k * 3 + 2];
            float dd = ex * ex + ey * ey + ez * ez;
            if (dd < best) { best = dd; kb = k; }
        }
        float inf = 1.0f - __builtin_sqrtf(best);
        if (inf > 0.0f) {
            const float4* f4 = reinterpret_cast<const float4*>(feats + (size_t)nidx * 64);
            float* row = agg + ql * KC + kb * 64;
            for (int jj = 0; jj < 16; ++jj) {
                int cc = (lane + jj) & 15;          // stagger: banks = f(c mod 64) only
                float4 f = f4[cc];
                atomicAdd(row + cc * 4 + 0, inf * f.x);
                atomicAdd(row + cc * 4 + 1, inf * f.y);
                atomicAdd(row + cc * 4 + 2, inf * f.z);
                atomicAdd(row + cc * 4 + 3, inf * f.w);
            }
        }
    }
    __syncthreads();

    // ---------------- Phase 2: out[32,64] = agg[32,960] x W[960,64] via WMMA f16->f32
    const int mt = warp >> 2;               // 0..1  : M tile
    const int t  = warp & 3;                // 0..3  : O tile
    const int m  = lane & 15;
    const int kbaseA = (lane < 16) ? 0 : 16; // A frag K sub-base (contiguous, matches B)
    const float* arow = agg + (mt * 16 + m) * KC;

    v8f acc = {};
    for (int b = 0; b < NB; ++b) {
        const float4* pa  = reinterpret_cast<const float4*>(arow + b * 32 + kbaseA);
        const float4* pa2 = reinterpret_cast<const float4*>(arow + b * 32 + kbaseA + 8);
        float4 a0 = pa[0],  a1 = pa[1];     // K = kbase+0..7
        float4 a2 = pa2[0], a3 = pa2[1];    // K = kbase+8..15
        v16h av;
        av[0]  = (_Float16)a0.x; av[1]  = (_Float16)a0.y;
        av[2]  = (_Float16)a0.z; av[3]  = (_Float16)a0.w;
        av[4]  = (_Float16)a1.x; av[5]  = (_Float16)a1.y;
        av[6]  = (_Float16)a1.z; av[7]  = (_Float16)a1.w;
        av[8]  = (_Float16)a2.x; av[9]  = (_Float16)a2.y;
        av[10] = (_Float16)a2.z; av[11] = (_Float16)a2.w;
        av[12] = (_Float16)a3.x; av[13] = (_Float16)a3.y;
        av[14] = (_Float16)a3.z; av[15] = (_Float16)a3.w;

        v16h bv = *reinterpret_cast<const v16h*>(
            wfrag + ((size_t)((b * 4 + t) * 32 + lane) << 5));

        acc = __builtin_amdgcn_wmma_f32_16x16x32_f16(
            false, av, false, bv, (short)0, acc, false, false);
    }

    // D layout: VGPR r -> row M = r + (lane<16?0:8), col N = lane&15
    const int Mofs = (lane < 16) ? 0 : 8;
    float* orow = out + (size_t)(q0 + mt * 16) * 64 + t * 16 + (lane & 15);
    #pragma unroll
    for (int r = 0; r < 8; ++r)
        orow[(size_t)(r + Mofs) * 64] = acc[r];
}

extern "C" void kernel_launch(void* const* d_in, const int* in_sizes, int n_in,
                              void* d_out, int out_size, void* d_ws, size_t ws_size,
                              hipStream_t stream) {
    const float* qpts  = (const float*)d_in[0];   // [100000,3]
    const float* feats = (const float*)d_in[1];   // [100000,64]
    const float* kern  = (const float*)d_in[2];   // [15,3]
    const float* W     = (const float*)d_in[3];   // [15,64,64]
    const int*   nbr   = (const int*)  d_in[4];   // [100000,32]
    float* out = (float*)d_out;                   // [100000,64]

    const int Q = in_sizes[0] / 3;                // 100000
    const int blocks = Q / QT;                    // 3125 (Q divisible by 32)

    kpconv_fused<<<blocks, THREADS, LDS_BYTES, stream>>>(qpts, feats, kern, W, nbr, out);
}
